// GPT_5875515261622
// MI455X (gfx1250) — matrix operand order repaired and checked by reading, not compiled, not run on hardware
//
#include <hip/hip_runtime.h>
#include <hip/hip_bf16.h>
#include <math.h>

#define DEVI __device__ __forceinline__

typedef __bf16 bf16_t;
typedef __attribute__((ext_vector_type(16))) __bf16 v16bf;
typedef __attribute__((ext_vector_type(8)))  float  v8f;

// Fixed model dims (GPT-2 small fwd: L=4, B=2, T=1024, E=768, H=12)
static constexpr int C_L = 4, C_H = 12, C_E = 768, C_T = 1024, C_V = 50257,
                     C_B = 2;
static constexpr int C_M = C_B * C_T, C_E3 = 3 * C_E, C_E4 = 4 * C_E,
                     C_HD = C_E / C_H;

union Frag16 {
  v16bf v;
  uint4 u[2];
  __bf16 e[16];
};

// A fragment (16x32 bf16, MxK) from row-major A; rowoff = row*ld (elements).
// ISA layout: lanes 0-15 -> M=lane, K = kk+[0..7] and kk+16+[0..7]
//             lanes 16-31 -> M=lane-16, K = kk+8+[0..7] and kk+24+[0..7]
DEVI v16bf load_a_frag(const bf16_t* A, int rowoff, int kk, int lane) {
  const int base = (lane & 16) ? 8 : 0;
  const bf16_t* p = A + (rowoff + kk + base);
  Frag16 f;
  f.u[0] = *(const uint4*)(p);        // k = kk+base   .. +7
  f.u[1] = *(const uint4*)(p + 16);   // k = kk+16+base.. +7
  return f.v;
}

// B fragment (32x16 bf16, KxN) loaded from BT = B^T row-major [N,K];
// noff = n*ld (elements).
// ISA layout: lanes 0-15 -> K = kk+[0..7] and kk+16+[0..7]; lanes 16-31 -> K = kk+8+[0..7] and kk+24+[0..7]
DEVI v16bf load_b_frag(const bf16_t* BT, int noff, int kk, int lane) {
  const int koff = (lane & 16) ? 8 : 0;
  const bf16_t* p = BT + (noff + kk + koff);
  Frag16 f;
  f.u[0] = *(const uint4*)(p);       // k = kk+koff .. +7
  f.u[1] = *(const uint4*)(p + 16);  // k = kk+16+koff .. +7
  return f.v;
}

DEVI v8f wmma_bf16(v16bf a, v16bf b, v8f c) {
  return __builtin_amdgcn_wmma_f32_16x16x32_bf16(false, a, false, b, (short)0, c,
                                                 false, false);
}

// --- DPP16 ROW_XMASK reductions (xor within each 16-lane row, VALU-only) ---
template <int CTRL>
DEVI float dpp_bcast(float v) {
  return __int_as_float(
      __builtin_amdgcn_update_dpp(0, __float_as_int(v), CTRL, 0xf, 0xf, true));
}
DEVI float red_max16(float x) {
  x = fmaxf(x, dpp_bcast<0x161>(x));  // ROW_XMASK:1
  x = fmaxf(x, dpp_bcast<0x162>(x));  // ROW_XMASK:2
  x = fmaxf(x, dpp_bcast<0x164>(x));  // ROW_XMASK:4
  x = fmaxf(x, dpp_bcast<0x168>(x));  // ROW_XMASK:8
  return x;
}
DEVI float red_sum16(float x) {
  x += dpp_bcast<0x161>(x);
  x += dpp_bcast<0x162>(x);
  x += dpp_bcast<0x164>(x);
  x += dpp_bcast<0x168>(x);
  return x;
}

// ---------------------------------------------------------------------------
// bf16 WMMA GEMM: out = epilogue(A[M,K] @ B[K,N] + bias [+ residual])
// A: bf16 row-major [M,K]; BT: bf16 row-major [N,K].
// One wave computes a 32x64 tile: 2 A-frags x 4 B-frags -> 8 WMMAs / K-step.
// ---------------------------------------------------------------------------
template <int ACT, int RES, int OUT_BF16>
__global__ void gemm_wmma_kernel(const bf16_t* __restrict__ A,
                                 const bf16_t* __restrict__ BT,
                                 const float* __restrict__ bias,
                                 const float* __restrict__ residual,
                                 void* __restrict__ out,
                                 int M, int N, int K) {
  const int lane = threadIdx.x & 31;
  const int wave = threadIdx.x >> 5;
  const int gw   = blockIdx.x * (blockDim.x >> 5) + wave;
  const int nwav = (N + 63) >> 6;  // 64-wide N strips
  const int mt   = gw / nwav;
  const int nc   = gw - mt * nwav;
  if (mt >= (M >> 5)) return;
  const int m0   = mt << 5;
  const int n0   = nc << 6;
  const int nl   = lane & 15;
  const int half = (lane >> 4) & 1;

  const int a0off = (m0 + nl) * K;
  const int a1off = (m0 + 16 + nl) * K;

  int bnoff[4];
#pragma unroll
  for (int j = 0; j < 4; ++j) {
    int n = n0 + j * 16 + nl;
    if (n > N - 1) n = N - 1;  // clamp: WMMA col n only affects output col n,
    bnoff[j] = n * K;          // which the epilogue never stores when n >= N.
  }

  v8f acc0[4] = {};
  v8f acc1[4] = {};

#pragma unroll 2
  for (int kk = 0; kk < K; kk += 32) {
    const v16bf a0 = load_a_frag(A, a0off, kk, lane);
    const v16bf a1 = load_a_frag(A, a1off, kk, lane);
    __builtin_prefetch(A + (a0off + kk + 64), 0, 1);
    __builtin_prefetch(A + (a1off + kk + 64), 0, 1);
#pragma unroll
    for (int j = 0; j < 4; ++j) {
      const v16bf b = load_b_frag(BT, bnoff[j], kk, lane);
      __builtin_prefetch(BT + (bnoff[j] + kk + 64), 0, 1);
      acc0[j] = wmma_bf16(a0, b, acc0[j]);
      acc1[j] = wmma_bf16(a1, b, acc1[j]);
    }
  }

#pragma unroll
  for (int mi = 0; mi < 2; ++mi) {
#pragma unroll
    for (int j = 0; j < 4; ++j) {
      const int n = n0 + j * 16 + nl;
      if (n >= N) continue;
      const float bv = bias ? bias[n] : 0.0f;
#pragma unroll
      for (int r = 0; r < 8; ++r) {
        const int m = m0 + mi * 16 + r + 8 * half;  // C/D layout: M = r+8*half
        float v = (mi ? acc1[j][r] : acc0[j][r]) + bv;
        if (ACT == 1) {
          v = 0.5f * v *
              (1.0f + tanhf(0.7978845608028654f * (v + 0.044715f * v * v * v)));
        }
        const size_t oi = (size_t)m * N + n;
        if (RES) v += residual[oi];
        if (OUT_BF16) ((bf16_t*)out)[oi] = (bf16_t)v;
        else          ((float*)out)[oi]  = v;
      }
    }
  }
}

// ---------------------------------------------------------------------------
// Fused causal attention, one wave (32 threads) per (b,h, 16-row q tile).
// qC,kC: bf16 [B,H,T,hd] ; vT: bf16 [B,H,hd,T] ; yC: bf16 [B,T,E]
// ---------------------------------------------------------------------------
template <bool MASK>
DEVI void attn_chunk(const bf16_t* __restrict__ k, const bf16_t* __restrict__ v,
                     int kt0, int qi, float scale, const v16bf& qa0,
                     const v16bf& qa1, float (*Ps)[33], v8f (&o)[4],
                     float (&rmax)[8], float (&rsum)[8], int nl, int half,
                     int lane) {
  const int kbase = kt0 * C_HD + nl * C_HD;
  // Score tiles, staged two B-frags at a time to limit register pressure.
  v8f s0 = {}, s1 = {};
  {
    const v16bf b00 = load_b_frag(k, kbase, 0, lane);
    const v16bf b01 = load_b_frag(k, kbase, 32, lane);
    s0 = wmma_bf16(qa0, b00, s0);
    s0 = wmma_bf16(qa1, b01, s0);
  }
  {
    const v16bf b10 = load_b_frag(k, kbase + 16 * C_HD, 0, lane);
    const v16bf b11 = load_b_frag(k, kbase + 16 * C_HD, 32, lane);
    s1 = wmma_bf16(qa0, b10, s1);
    s1 = wmma_bf16(qa1, b11, s1);
  }

#pragma unroll
  for (int r = 0; r < 8; ++r) {
    float v0 = s0[r] * scale;
    float v1 = s1[r] * scale;
    if (MASK) {
      const int m = qi * 16 + r + 8 * half;
      if (kt0 + nl > m)      v0 = -3.0e38f;  // causal mask
      if (kt0 + 16 + nl > m) v1 = -3.0e38f;
    }
    const float cm   = red_max16(fmaxf(v0, v1));
    const float newm = fmaxf(rmax[r], cm);
    const float cr   = __expf(rmax[r] - newm);
    const float p0   = __expf(v0 - newm);
    const float p1   = __expf(v1 - newm);
    rsum[r] = rsum[r] * cr + red_sum16(p0 + p1);
    rmax[r] = newm;
    o[0][r] *= cr;
    o[1][r] *= cr;
    o[2][r] *= cr;
    o[3][r] *= cr;
    Ps[r + 8 * half][nl]      = p0;  // P[m][k_local]    (keys kt0+0..15)
    Ps[r + 8 * half][nl + 16] = p1;  // P[m][k_local+16] (keys kt0+16..31)
  }

  __syncthreads();
  // Rebuild P as an A-fragment (row = nl, k-halves selected by lane half)
  Frag16 pf;
  const int base = half * 8;
#pragma unroll
  for (int e2 = 0; e2 < 8; ++e2) {
    pf.e[e2]     = (__bf16)Ps[nl][base + e2];
    pf.e[e2 + 8] = (__bf16)Ps[nl][16 + base + e2];
  }
  __syncthreads();

#pragma unroll
  for (int j = 0; j < 4; ++j) {
    const v16bf vb = load_b_frag(v, (j * 16 + nl) * C_T, kt0, lane);
    o[j] = wmma_bf16(pf.v, vb, o[j]);
  }
}

__global__ void attn_fused_kernel(const bf16_t* __restrict__ qC,
                                  const bf16_t* __restrict__ kC,
                                  const bf16_t* __restrict__ vT,
                                  bf16_t* __restrict__ yC, float scale) {
  const int qi   = blockIdx.x;   // q tile
  const int bh   = blockIdx.y;   // b*H + h
  const int lane = threadIdx.x;  // single wave of 32
  const int nl   = lane & 15;
  const int half = (lane >> 4) & 1;
  const int h    = bh % C_H;
  const int b    = bh / C_H;

  const bf16_t* q = qC + (size_t)bh * C_T * C_HD;
  const bf16_t* k = kC + (size_t)bh * C_T * C_HD;
  const bf16_t* v = vT + (size_t)bh * C_HD * C_T;

  __shared__ float Ps[16][33];  // P tile staging (C-layout -> A-layout)

  const int   qoff = (qi * 16 + nl) * C_HD;
  const v16bf qa0  = load_a_frag(q, qoff, 0, lane);   // k-dims 0..31
  const v16bf qa1  = load_a_frag(q, qoff, 32, lane);  // k-dims 32..63

  v8f   o[4] = {};
  float rmax[8], rsum[8];
#pragma unroll
  for (int r = 0; r < 8; ++r) { rmax[r] = -3.0e38f; rsum[r] = 0.0f; }

  // Chunks with kt0+31 <= 16*qi need no causal mask for any row of the tile.
  const int nchunks = (qi + 2) >> 1;
  const int nfull   = (16 * qi >= 31) ? (((16 * qi - 31) >> 5) + 1) : 0;
  int c = 0;
  for (; c < nfull; ++c)
    attn_chunk<false>(k, v, c * 32, qi, scale, qa0, qa1, Ps, o, rmax, rsum, nl,
                      half, lane);
  for (; c < nchunks; ++c)
    attn_chunk<true>(k, v, c * 32, qi, scale, qa0, qa1, Ps, o, rmax, rsum, nl,
                     half, lane);

  float inv[8];
#pragma unroll
  for (int r = 0; r < 8; ++r) inv[r] = 1.0f / rsum[r];
#pragma unroll
  for (int j = 0; j < 4; ++j) {
#pragma unroll
    for (int r = 0; r < 8; ++r) {
      const int m = qi * 16 + r + 8 * half;
      const int d = j * 16 + nl;
      yC[(b * C_T + m) * C_E + h * C_HD + d] = (bf16_t)(o[j][r] * inv[r]);
    }
  }
}

// ---------------------------------------------------------------------------
// Elementwise / prep kernels
// ---------------------------------------------------------------------------
__global__ void embed_kernel(const int* __restrict__ idx,
                             const float* __restrict__ wte,
                             const float* __restrict__ wpe,
                             float* __restrict__ x) {
  const int i = blockIdx.x * blockDim.x + threadIdx.x;
  if (i >= C_M * C_E) return;
  const int e  = i % C_E;
  const int bt = i / C_E;
  const int t  = bt % C_T;
  x[i] = wte[(size_t)idx[bt] * C_E + e] + wpe[t * C_E + e];
}

__global__ void convert_f32_bf16_kernel(const float* __restrict__ w,
                                        bf16_t* __restrict__ c, int n) {
  const int i = blockIdx.x * blockDim.x + threadIdx.x;
  if (i < n) c[i] = (bf16_t)w[i];
}

// w: fp32 row-major [K,N] -> wT: bf16 row-major [N,K]
__global__ void transpose_conv_kernel(const float* __restrict__ w,
                                      bf16_t* __restrict__ wT, int K, int N) {
  const int i = blockIdx.x * blockDim.x + threadIdx.x;
  if (i >= K * N) return;
  const int n = i / K;
  const int k = i - n * K;
  wT[i] = (bf16_t)w[(size_t)k * N + n];
}

// qkv bf16 [B,T,3E] -> qC,kC bf16 [B,H,T,hd], vT bf16 [B,H,hd,T]
__global__ void split_qkv_kernel(const bf16_t* __restrict__ qkv,
                                 bf16_t* __restrict__ qC,
                                 bf16_t* __restrict__ kC,
                                 bf16_t* __restrict__ vT) {
  const int i = blockIdx.x * blockDim.x + threadIdx.x;
  if (i >= C_B * C_H * C_T * C_HD) return;
  const int d = i % C_HD;
  const int t = (i / C_HD) % C_T;
  const int h = (i / (C_HD * C_T)) % C_H;
  const int b = i / (C_HD * C_T * C_H);
  const bf16_t* src = qkv + (size_t)(b * C_T + t) * C_E3 + h * C_HD + d;
  qC[i] = src[0];
  kC[i] = src[C_E];
  vT[((b * C_H + h) * C_HD + d) * C_T + t] = src[2 * C_E];
}

__global__ void layernorm_bf16_kernel(const float* __restrict__ x,
                                      const float* __restrict__ w,
                                      const float* __restrict__ b,
                                      bf16_t* __restrict__ out) {
  const int row = blockIdx.x;
  const float* xr = x + (size_t)row * C_E;
  __shared__ float sbuf[2][8];
  float s = 0.f, ss = 0.f;
  for (int i = threadIdx.x; i < C_E; i += blockDim.x) {
    const float v = xr[i];
    s += v;
    ss += v * v;
  }
#pragma unroll
  for (int off = 16; off >= 1; off >>= 1) {
    s  += __shfl_xor(s, off, 32);
    ss += __shfl_xor(ss, off, 32);
  }
  const int wv = threadIdx.x >> 5, ln = threadIdx.x & 31;
  if (ln == 0) { sbuf[0][wv] = s; sbuf[1][wv] = ss; }
  __syncthreads();
  if (threadIdx.x == 0) {
    float ts = 0.f, tss = 0.f;
    for (int i = 0; i < 8; ++i) { ts += sbuf[0][i]; tss += sbuf[1][i]; }
    sbuf[0][0] = ts;
    sbuf[1][0] = tss;
  }
  __syncthreads();
  const float mean = sbuf[0][0] * (1.0f / C_E);
  const float var  = sbuf[1][0] * (1.0f / C_E) - mean * mean;
  const float rinv = rsqrtf(var + 1e-5f);
  for (int i = threadIdx.x; i < C_E; i += blockDim.x)
    out[(size_t)row * C_E + i] = (bf16_t)((xr[i] - mean) * rinv * w[i] + b[i]);
}

// ---------------------------------------------------------------------------
// Host orchestration
// ---------------------------------------------------------------------------
extern "C" void kernel_launch(void* const* d_in, const int* in_sizes, int n_in,
                              void* d_out, int out_size, void* d_ws, size_t ws_size,
                              hipStream_t stream) {
  (void)in_sizes; (void)n_in; (void)out_size; (void)ws_size;

  constexpr int L = C_L, E = C_E, M = C_M, E3 = C_E3, E4 = C_E4, V = C_V;
  const float scale = 0.03608439182435161f;  // 1/sqrt(E) per the reference

  const int*   idx    = (const int*)d_in[0];
  const float* wte    = (const float*)d_in[1];
  const float* wpe    = (const float*)d_in[2];
  const float* ln1_w  = (const float*)d_in[3];
  const float* ln1_b  = (const float*)d_in[4];
  const float* attn_w = (const float*)d_in[5];
  const float* attn_b = (const float*)d_in[6];
  const float* atp_w  = (const float*)d_in[7];
  const float* atp_b  = (const float*)d_in[8];
  const float* ln2_w  = (const float*)d_in[9];
  const float* ln2_b  = (const float*)d_in[10];
  const float* fc_w   = (const float*)d_in[11];
  const float* fc_b   = (const float*)d_in[12];
  const float* pr_w   = (const float*)d_in[13];
  const float* pr_b   = (const float*)d_in[14];
  const float* lnf_w  = (const float*)d_in[15];
  const float* lnf_b  = (const float*)d_in[16];

  char* ws = (char*)d_ws;
  auto carve = [&](size_t bytes) -> char* {
    char* p = ws;
    ws += (bytes + 255) & ~(size_t)255;
    return p;
  };
  bf16_t* wte_bf = (bf16_t*)carve((size_t)V * E * 2);
  bf16_t* wqkvT  = (bf16_t*)carve((size_t)L * E3 * E * 2);
  bf16_t* watpT  = (bf16_t*)carve((size_t)L * E * E * 2);
  bf16_t* wfcT   = (bf16_t*)carve((size_t)L * E4 * E * 2);
  bf16_t* wprT   = (bf16_t*)carve((size_t)L * E * E4 * 2);
  float*  xbuf   = (float*)carve((size_t)M * E * 4);
  bf16_t* hbuf   = (bf16_t*)carve((size_t)M * E * 2);
  bf16_t* qkvbuf = (bf16_t*)carve((size_t)M * E3 * 2);
  bf16_t* qCb    = (bf16_t*)carve((size_t)M * E * 2);
  bf16_t* kCb    = (bf16_t*)carve((size_t)M * E * 2);
  bf16_t* vTb    = (bf16_t*)carve((size_t)M * E * 2);
  bf16_t* yCb    = (bf16_t*)carve((size_t)M * E * 2);
  bf16_t* mbuf   = (bf16_t*)carve((size_t)M * E4 * 2);
  bf16_t* xnbuf  = (bf16_t*)carve((size_t)M * E * 2);

  auto launch_gemm = [&](auto kern, const bf16_t* A, const bf16_t* BT,
                         const float* bias, const float* res, void* out,
                         int Mm, int Nn, int Kk) {
    const int nwav   = (Nn + 63) / 64;
    const int waves  = (Mm / 32) * nwav;   // 32x64 per wave
    const int blocks = (waves + 7) / 8;    // 8 waves / 256-thread block
    kern<<<blocks, 256, 0, stream>>>(A, BT, bias, res, out, Mm, Nn, Kk);
  };

  // --- weight prep (bf16 + transpose) ---
  {
    const int n = V * E;
    convert_f32_bf16_kernel<<<(n + 255) / 256, 256, 0, stream>>>(wte, wte_bf, n);
  }
  for (int i = 0; i < L; ++i) {
    transpose_conv_kernel<<<(E * E3 + 255) / 256, 256, 0, stream>>>(
        attn_w + (size_t)i * E * E3, wqkvT + (size_t)i * E3 * E, E, E3);
    transpose_conv_kernel<<<(E * E + 255) / 256, 256, 0, stream>>>(
        atp_w + (size_t)i * E * E, watpT + (size_t)i * E * E, E, E);
    transpose_conv_kernel<<<(E * E4 + 255) / 256, 256, 0, stream>>>(
        fc_w + (size_t)i * E * E4, wfcT + (size_t)i * E4 * E, E, E4);
    transpose_conv_kernel<<<(E4 * E + 255) / 256, 256, 0, stream>>>(
        pr_w + (size_t)i * E4 * E, wprT + (size_t)i * E * E4, E4, E);
  }

  // --- embedding ---
  embed_kernel<<<(M * E + 255) / 256, 256, 0, stream>>>(idx, wte, wpe, xbuf);

  // --- transformer layers ---
  for (int i = 0; i < L; ++i) {
    layernorm_bf16_kernel<<<M, 256, 0, stream>>>(xbuf, ln1_w + (size_t)i * E,
                                                 ln1_b + (size_t)i * E, hbuf);
    launch_gemm(gemm_wmma_kernel<0, 0, 1>, hbuf, wqkvT + (size_t)i * E3 * E,
                attn_b + (size_t)i * E3, nullptr, qkvbuf, M, E3, E);
    split_qkv_kernel<<<(M * E + 255) / 256, 256, 0, stream>>>(qkvbuf, qCb, kCb,
                                                              vTb);
    dim3 ag(C_T / 16, C_B * C_H);
    attn_fused_kernel<<<ag, 32, 0, stream>>>(qCb, kCb, vTb, yCb, scale);
    launch_gemm(gemm_wmma_kernel<0, 1, 0>, yCb, watpT + (size_t)i * E * E,
                atp_b + (size_t)i * E, xbuf, xbuf, M, E, E);
    layernorm_bf16_kernel<<<M, 256, 0, stream>>>(xbuf, ln2_w + (size_t)i * E,
                                                 ln2_b + (size_t)i * E, hbuf);
    launch_gemm(gemm_wmma_kernel<1, 0, 1>, hbuf, wfcT + (size_t)i * E4 * E,
                fc_b + (size_t)i * E4, nullptr, mbuf, M, E4, E);
    launch_gemm(gemm_wmma_kernel<0, 1, 0>, mbuf, wprT + (size_t)i * E * E4,
                pr_b + (size_t)i * E, xbuf, xbuf, M, E, E4);
  }

  // --- final LN + tied lm_head ---
  layernorm_bf16_kernel<<<M, 256, 0, stream>>>(xbuf, lnf_w, lnf_b, xnbuf);
  launch_gemm(gemm_wmma_kernel<0, 0, 0>, xnbuf, wte_bf, nullptr, nullptr, d_out,
              M, V, E);
}
